// VectorQuantizer_60954175864979
// MI455X (gfx1250) — matrix operand order repaired and checked by reading, compiled, not run on hardware
//
#include <hip/hip_runtime.h>

typedef __attribute__((ext_vector_type(16))) _Float16 v16h;
typedef __attribute__((ext_vector_type(8)))  float    v8f;

#define K_CODES 1024
#define D_DIM 64
#define ROWS_PER_BLOCK 128
#define THREADS 256
#define N_TILES (K_CODES / 16)                // 64 column tiles
#define B_FRAG_ELEMS (N_TILES * 2 * 32 * 16)  // 65536 f16 = 128 KB

__global__ __launch_bounds__(THREADS) void vq_wmma_kernel(
    const float* __restrict__ x,
    const float* __restrict__ emb,
    float* __restrict__ out)
{
    // B fragments (f16, WMMA-fragment-ordered), per-column bias, winning indices
    __shared__ _Float16 sB[B_FRAG_ELEMS];       // 128 KB
    __shared__ float    sBias[K_CODES];         // 4 KB
    __shared__ int      sIdx[ROWS_PER_BLOCK];   // 512 B

    const int tid  = threadIdx.x;
    const int lane = tid & 31;
    const int wave = tid >> 5;      // 0..7, one 16-row M tile per wave
    const int half = lane >> 4;     // 0 or 1
    const int l16  = lane & 15;

    // ---- Stage embeddings into LDS in WMMA B-fragment order ----
    // Consumption layout: v16h element e of lane n for tile (t,kb) =
    //   E[kb*32 + (n<16 ? e : 16+e)][t*16 + (n&15)]
    // Source-major iteration: coalesced float4 global reads, scattered f16 LDS writes.
    const float4* emb4 = (const float4*)emb;
    for (int q4 = tid; q4 < (D_DIM * K_CODES) / 4; q4 += THREADS) {
        const int q   = q4 << 2;         // flat f32 index into emb
        const int d   = q >> 10;         // feature row 0..63
        const int c0  = q & 1023;        // column base (4 consecutive)
        const float4 v = emb4[q4];
        const int kb  = d >> 5;          // K-block (0: K 0..31, 1: K 32..63)
        const int r   = d & 31;
        const int e   = r & 15;
        const int lnB = (r < 16) ? 0 : 16;
        const float vv[4] = {v.x, v.y, v.z, v.w};
        #pragma unroll
        for (int j = 0; j < 4; ++j) {
            const int c  = c0 + j;
            const int t  = c >> 4;
            const int ln = lnB + (c & 15);
            sB[((t * 2 + kb) * 32 + ln) * 16 + e] = (_Float16)vv[j];
        }
    }
    // ---- bias[c] = 0.5 * ||e_c||^2 (coalesced across threads per d) ----
    for (int c = tid; c < K_CODES; c += THREADS) {
        float acc = 0.0f;
        #pragma unroll 8
        for (int d = 0; d < D_DIM; ++d) {
            const float v = emb[d * K_CODES + c];
            acc += v * v;
        }
        sBias[c] = 0.5f * acc;
    }
    __syncthreads();

    // ---- Load this wave's 16x64 A tile (f32 -> f16) in WMMA A-fragment order ----
    // A element e of lane L: row = L&15, K = e + (L<16 ? 0 : 16); a1: +32
    const long rowA = (long)blockIdx.x * ROWS_PER_BLOCK + wave * 16 + l16;
    const float4* xr4 = (const float4*)(x + rowA * D_DIM);
    const int q0 = half * 4;            // k0/4, k0 = half*16
    float4 ra[8];
    #pragma unroll
    for (int rr = 0; rr < 4; ++rr) {    // runs at K = k0 + {0,8,32,40}
        ra[rr * 2 + 0] = xr4[q0 + (rr >> 1) * 8 + (rr & 1) * 2 + 0];
        ra[rr * 2 + 1] = xr4[q0 + (rr >> 1) * 8 + (rr & 1) * 2 + 1];
    }
    v16h a0, a1;
    #pragma unroll
    for (int rr = 0; rr < 2; ++rr) {    // rr=0 -> a0 (K 0..31), rr=1 -> a1 (K 32..63)
        v16h& a = (rr == 0) ? a0 : a1;
        const float4 u0 = ra[rr * 4 + 0], u1 = ra[rr * 4 + 1];
        const float4 u2 = ra[rr * 4 + 2], u3 = ra[rr * 4 + 3];
        a[0]  = (_Float16)u0.x; a[1]  = (_Float16)u0.y;
        a[2]  = (_Float16)u0.z; a[3]  = (_Float16)u0.w;
        a[4]  = (_Float16)u1.x; a[5]  = (_Float16)u1.y;
        a[6]  = (_Float16)u1.z; a[7]  = (_Float16)u1.w;
        a[8]  = (_Float16)u2.x; a[9]  = (_Float16)u2.y;
        a[10] = (_Float16)u2.z; a[11] = (_Float16)u2.w;
        a[12] = (_Float16)u3.x; a[13] = (_Float16)u3.y;
        a[14] = (_Float16)u3.z; a[15] = (_Float16)u3.w;
    }

    float bestv[8];
    int   besti[8];
    #pragma unroll
    for (int i = 0; i < 8; ++i) { bestv[i] = -3.402823466e38f; besti[i] = 0; }

    // ---- Main loop: 64 column tiles, 2 WMMAs each (D=64 -> two K=32 steps) ----
    const v16h* sBv = (const v16h*)sB;
    #pragma unroll 4
    for (int t = 0; t < N_TILES; ++t) {
        const v16h b0 = sBv[(t * 2 + 0) * 32 + lane];
        const v16h b1 = sBv[(t * 2 + 1) * 32 + lane];
        v8f c = {0.f, 0.f, 0.f, 0.f, 0.f, 0.f, 0.f, 0.f};
        c = __builtin_amdgcn_wmma_f32_16x16x32_f16(false, a0, false, b0,
                                                   (short)0, c, false, false);
        c = __builtin_amdgcn_wmma_f32_16x16x32_f16(false, a1, false, b1,
                                                   (short)0, c, false, false);
        const int   col  = t * 16 + l16;
        const float bias = sBias[col];
        // C layout: VGPR i, lanes 0-15 -> row i, lanes 16-31 -> row 8+i, col = lane&15
        #pragma unroll
        for (int i = 0; i < 8; ++i) {
            const float s = c[i] - bias;   // score = z.e - 0.5||e||^2 (maximize)
            if (s > bestv[i]) { bestv[i] = s; besti[i] = col; }
        }
    }

    // ---- Argmax reduction across the 16 lanes of each half-wave ----
    #pragma unroll
    for (int m = 1; m <= 8; m <<= 1) {
        #pragma unroll
        for (int i = 0; i < 8; ++i) {
            const float ov = __shfl_xor(bestv[i], m, 32);
            const int   oi = __shfl_xor(besti[i], m, 32);
            if (ov > bestv[i] || (ov == bestv[i] && oi < besti[i])) {
                bestv[i] = ov; besti[i] = oi;
            }
        }
    }
    if (l16 == 0) {
        #pragma unroll
        for (int i = 0; i < 8; ++i)
            sIdx[wave * 16 + half * 8 + i] = besti[i];
    }
    __syncthreads();

    // ---- Gather winning codewords to output (coalesced stores) ----
    float* outB = out + (long)blockIdx.x * ROWS_PER_BLOCK * D_DIM;
    for (int e2 = tid; e2 < ROWS_PER_BLOCK * D_DIM; e2 += THREADS) {
        const int r = e2 >> 6;
        const int d = e2 & 63;
        outB[e2] = emb[d * K_CODES + sIdx[r]];
    }
}

extern "C" void kernel_launch(void* const* d_in, const int* in_sizes, int n_in,
                              void* d_out, int out_size, void* d_ws, size_t ws_size,
                              hipStream_t stream) {
    (void)n_in; (void)out_size; (void)d_ws; (void)ws_size;
    const float* x   = (const float*)d_in[0];   // [16*64*64, 64] f32
    const float* emb = (const float*)d_in[1];   // [64, 1024] f32
    float* out = (float*)d_out;                 // same shape as x

    const int n_rows = in_sizes[0] / D_DIM;           // 65536
    const int blocks = n_rows / ROWS_PER_BLOCK;       // 512
    vq_wmma_kernel<<<blocks, THREADS, 0, stream>>>(x, emb, out);
}
